// KHopfield_25434796327560
// MI455X (gfx1250) — matrix operand order repaired and checked by reading, not compiled, not run on hardware
//
#include <hip/hip_runtime.h>
#include <hip/hip_bf16.h>
#include <float.h>

typedef __attribute__((ext_vector_type(16))) _Float16 v16h;
typedef __attribute__((ext_vector_type(8)))  _Float16 v8h;
typedef __attribute__((ext_vector_type(8)))  float    v8f;

static constexpr int B = 512;
static constexpr int M = 65536;
static constexpr int NDIM = 64;
static constexpr int K = 4;

// ---------------------------------------------------------------------------
// Tensor Data Mover support (gfx1250).  Probe-verified builtin:
//   ROCm 7.2 / clang-22:  5 args (uint32x4, int32x8, int32x4, int32x4, i32)
//   clang-23 toolchain:   6 args (uint32x4, int32x8, int32x4, int32x4, int32x8, i32)
// ---------------------------------------------------------------------------
#if defined(__gfx1250__) && __has_builtin(__builtin_amdgcn_tensor_load_to_lds) && \
    __has_builtin(__builtin_amdgcn_s_wait_tensorcnt)
#define HAVE_TDM 1
#else
#define HAVE_TDM 0
#endif

#if HAVE_TDM
typedef unsigned int v4u_ __attribute__((ext_vector_type(4)));
typedef int          v8i_ __attribute__((ext_vector_type(8)));
typedef int          v4i_ __attribute__((ext_vector_type(4)));

// Load a (rows x cols) f32 tile (row stride = `stride` elements) from global
// memory into LDS at byte offset lds_addr.  Descriptor per CDNA5 ISA §8.3-8.6.
__device__ __forceinline__ void tdm_load_f32(unsigned lds_addr, const void* gptr,
                                             unsigned cols, unsigned rows,
                                             unsigned stride) {
    unsigned long long ga = (unsigned long long)gptr;
    v4u_ g0;
    g0.x = 1u;                                              // count=1, user desc
    g0.y = lds_addr;                                        // lds_addr
    g0.z = (unsigned)(ga & 0xFFFFFFFFu);                    // global_addr[31:0]
    g0.w = (unsigned)((ga >> 32) & 0x01FFFFFFu) | (2u << 30); // ga[56:32] | type=2
    v8i_ g1;
    g1[0] = (int)(2u << 16);                                // data_size=2 (4 bytes)
    g1[1] = (int)((cols & 0xFFFFu) << 16);                  // tensor_dim0[15:0]
    g1[2] = (int)(((cols >> 16) & 0xFFFFu) |                // tensor_dim0[31:16]
                  ((rows & 0xFFFFu) << 16));                // tensor_dim1[15:0]
    g1[3] = (int)(((rows >> 16) & 0xFFFFu) |                // tensor_dim1[31:16]
                  ((cols & 0xFFFFu) << 16));                // tile_dim0
    g1[4] = (int)(rows & 0xFFFFu);                          // tile_dim1 (tile_dim2=0)
    g1[5] = (int)stride;                                    // tensor_dim0_stride[31:0]
    g1[6] = 0;                                              // stride hi, dim1_stride lo
    g1[7] = 0;
    v4i_ z4 = {0, 0, 0, 0};
#if __clang_major__ >= 23
    v8i_ z8 = {0, 0, 0, 0, 0, 0, 0, 0};
    __builtin_amdgcn_tensor_load_to_lds(g0, g1, z4, z4, z8, 0);
#else
    __builtin_amdgcn_tensor_load_to_lds(g0, g1, z4, z4, 0);
#endif
}

__device__ __forceinline__ unsigned lds_offset_of(const void* p) {
    // Flat shared-aperture addresses carry the LDS byte offset in the low 32 bits
    return (unsigned)(unsigned long long)(size_t)p;
}
#endif

__device__ __forceinline__ float fast_sigmoid(float t) {
    return __builtin_amdgcn_rcpf(1.0f + __expf(-t));   // v_exp_f32 + v_rcp_f32
}

// Insert value v into descending-sorted 5-array (branchless, const indices)
__device__ __forceinline__ void insert5(float (&a)[5], float v) {
#pragma unroll
    for (int i = 0; i < 5; ++i) {
        float old = a[i];
        bool g = v > old;
        a[i] = g ? v : old;
        v    = g ? old : v;
    }
}

// ---------------------------------------------------------------------------
// K0: row squared norms for memories (M rows) and x (B rows)
// ---------------------------------------------------------------------------
__global__ void norms_kernel(const float* __restrict__ x,
                             const float* __restrict__ mem,
                             float* __restrict__ xn,
                             float* __restrict__ mn) {
    int i = blockIdx.x * blockDim.x + threadIdx.x;
    const float4* row = nullptr;
    if (i < M) {
        row = (const float4*)(mem + (size_t)i * NDIM);
    } else if (i < M + B) {
        row = (const float4*)(x + (size_t)(i - M) * NDIM);
    } else {
        return;
    }
    float s = 0.0f;
#pragma unroll
    for (int q = 0; q < NDIM / 4; ++q) {
        float4 v = row[q];
        s += v.x * v.x + v.y * v.y + v.z * v.z + v.w * v.w;
    }
    if (i < M) mn[i] = s;
    else       xn[i - M] = s;
}

// ---------------------------------------------------------------------------
// K1: convert x and memories to f16 for WMMA
// ---------------------------------------------------------------------------
__global__ void tohalf_kernel(const float* __restrict__ x,
                              const float* __restrict__ mem,
                              _Float16* __restrict__ x16,
                              _Float16* __restrict__ m16) {
    int i = blockIdx.x * blockDim.x + threadIdx.x;
    const int ME = M * NDIM;
    const int XE = B * NDIM;
    if (i < ME) {
        m16[i] = (_Float16)mem[i];
    } else if (i < ME + XE) {
        x16[i - ME] = (_Float16)x[i - ME];
    }
}

// ---------------------------------------------------------------------------
// K2: WMMA GEMM  S = x @ mem^T (512x65536, K=64), fused epilogue:
//     negd[b,m] = min(2S - |x_b|^2 - |m_m|^2, 0)
// ---------------------------------------------------------------------------
__global__ void gemm_negd_kernel(const _Float16* __restrict__ x16,
                                 const _Float16* __restrict__ m16,
                                 const float* __restrict__ xn,
                                 const float* __restrict__ mn,
                                 float* __restrict__ negd) {
    const int lane = threadIdx.x & 31;
    const int wave = threadIdx.x >> 5;
    const int tile = blockIdx.x * 8 + wave;
    const int rowTile = tile & 31;
    const int colTile = tile >> 5;
    const int b0 = rowTile * 16;
    const int m0 = colTile * 16;
    const int r = lane & 15;
    const int h = lane >> 4;

    v8f c = {};
#pragma unroll
    for (int ks = 0; ks < 2; ++ks) {
        const _Float16* pa = x16 + (size_t)(b0 + r) * NDIM + ks * 32 + h * 16;
        v8h alo = *(const v8h*)pa;
        v8h ahi = *(const v8h*)(pa + 8);
        v16h a;
#pragma unroll
        for (int q = 0; q < 8; ++q) { a[q] = alo[q]; a[q + 8] = ahi[q]; }

        const _Float16* pb = m16 + (size_t)(m0 + r) * NDIM + ks * 32 + h * 16;
        v16h bfrag = *(const v16h*)pb;

        c = __builtin_amdgcn_wmma_f32_16x16x32_f16(
                false, a, false, bfrag, (short)0, c, false, false);
    }

    const int col = m0 + r;
    const float mnv = mn[col];
    const int rowBase = b0 + h * 8;
#pragma unroll
    for (int q = 0; q < 8; ++q) {
        const int row = rowBase + q;
        float s = c[q];
        float v = fminf(2.0f * s - xn[row] - mnv, 0.0f);
        negd[(size_t)row * M + col] = v;
    }
}

// ---------------------------------------------------------------------------
// K3: per-row top-5 + LML bisection (200 iters, N=1..4).
// One workgroup (1024 threads) per row; 256 KB row cached in LDS, staged by
// a single tensor_load_to_lds when the TDM builtin is available.
// ---------------------------------------------------------------------------
__global__ void lml_bisect_kernel(const float* __restrict__ negd,
                                  float* __restrict__ nu) {
    extern __shared__ float sm[];
    float* rowc  = sm;                  // 65536 floats
    float* tops  = sm + M;              // 32*5
    float* bcast = tops + 32 * 5;       // 5 (+pad)
    float* red   = bcast + 8;           // 32*4

    const int b    = blockIdx.x;
    const int tid  = threadIdx.x;
    const int lane = tid & 31;
    const int wave = tid >> 5;
    const float* g = negd + (size_t)b * M;

    float t5[5] = {-FLT_MAX, -FLT_MAX, -FLT_MAX, -FLT_MAX, -FLT_MAX};

#if HAVE_TDM
    // One TDM descriptor moves the whole 256 KB row (4 x 16384 f32 tile)
    if (tid == 0) {
        tdm_load_f32(lds_offset_of(rowc), g, 16384u, 4u, 16384u);
        __builtin_amdgcn_s_wait_tensorcnt(0);
    }
    __syncthreads();
    const float2* row2 = (const float2*)rowc;
#pragma unroll 4
    for (int i = 0; i < M / 2048; ++i) {
        float2 v = row2[tid + i * 1024];
        insert5(t5, v.x);
        insert5(t5, v.y);
    }
#else
#pragma unroll 4
    for (int i = 0; i < M / 1024; ++i) {
        int idx = tid + i * 1024;
        if (i + 1 < M / 1024) __builtin_prefetch(g + idx + 1024, 0, 0);
        float v = g[idx];
        rowc[idx] = v;
        insert5(t5, v);
    }
#endif

    // wave32 butterfly merge of sorted-5 lists
#pragma unroll
    for (int off = 16; off >= 1; off >>= 1) {
        float o[5];
#pragma unroll
        for (int q = 0; q < 5; ++q) o[q] = __shfl_xor(t5[q], off, 32);
#pragma unroll
        for (int q = 0; q < 5; ++q) insert5(t5, o[q]);
    }
    if (lane == 0) {
#pragma unroll
        for (int q = 0; q < 5; ++q) tops[wave * 5 + q] = t5[q];
    }
    __syncthreads();
    if (wave == 0) {
        float u5[5];
#pragma unroll
        for (int q = 0; q < 5; ++q) u5[q] = tops[lane * 5 + q];
#pragma unroll
        for (int off = 16; off >= 1; off >>= 1) {
            float o[5];
#pragma unroll
            for (int q = 0; q < 5; ++q) o[q] = __shfl_xor(u5[q], off, 32);
#pragma unroll
            for (int q = 0; q < 5; ++q) insert5(u5, o[q]);
        }
        if (lane == 0) {
#pragma unroll
            for (int q = 0; q < 5; ++q) bcast[q] = u5[q];
        }
    }
    __syncthreads();

    float top[5];
#pragma unroll
    for (int q = 0; q < 5; ++q) top[q] = bcast[q];

    float lo[K], hi[K];
#pragma unroll
    for (int j = 0; j < K; ++j) { lo[j] = -top[j] - 7.0f; hi[j] = -top[j + 1] + 7.0f; }

    const float2* row2h = (const float2*)rowc;
    const float4* red4  = (const float4*)red;

    for (int it = 0; it < 200; ++it) {
        float mid[K], s[K];
#pragma unroll
        for (int j = 0; j < K; ++j) { mid[j] = 0.5f * (lo[j] + hi[j]); s[j] = 0.0f; }
#pragma unroll 4
        for (int i = 0; i < M / 2048; ++i) {
            float2 v = row2h[tid + i * 1024];     // ds_load_b64, bank-clean
#pragma unroll
            for (int j = 0; j < K; ++j) {
                s[j] += fast_sigmoid(v.x + mid[j]);
                s[j] += fast_sigmoid(v.y + mid[j]);
            }
        }
#pragma unroll
        for (int off = 16; off >= 1; off >>= 1) {
#pragma unroll
            for (int j = 0; j < K; ++j) s[j] += __shfl_xor(s[j], off, 32);
        }
        if (lane == 0) {
#pragma unroll
            for (int j = 0; j < K; ++j) red[wave * K + j] = s[j];
        }
        __syncthreads();
        float f0 = 0.0f, f1 = 0.0f, f2 = 0.0f, f3 = 0.0f;
        for (int w = 0; w < 32; ++w) {
            float4 rv = red4[w];                  // ds_load_b128 broadcast
            f0 += rv.x; f1 += rv.y; f2 += rv.z; f3 += rv.w;
        }
        float f[K] = {f0, f1, f2, f3};
#pragma unroll
        for (int j = 0; j < K; ++j) {
            bool below = (f[j] - (float)(j + 1)) < 0.0f;
            lo[j] = below ? mid[j] : lo[j];
            hi[j] = below ? hi[j]  : mid[j];
        }
        __syncthreads();
    }
    if (tid == 0) {
#pragma unroll
        for (int j = 0; j < K; ++j) nu[b * K + j] = 0.5f * (lo[j] + hi[j]);
    }
}

// ---------------------------------------------------------------------------
// K4: r0[b,:,j] = sum_m w_j[m] * mem[m,:], w = sigmoid first-differences.
// One workgroup (256 threads) per b.  Memory tiles (256x64 f32 = 64 KB) are
// double-buffered through LDS via the TDM: tile t+1 streams in while tile t
// is consumed (s_wait_tensorcnt 1 overlap).
// ---------------------------------------------------------------------------
__global__ void ksm_gemv_kernel(const float* __restrict__ negd,
                                const float* __restrict__ nu,
                                const float* __restrict__ mem,
                                float* __restrict__ r0) {
    extern __shared__ float sm[];
    float* tiles = sm;                  // 2 x 256*64 floats
    float* wl    = sm + 2 * 256 * 64;   // 256*4 floats

    const int b   = blockIdx.x;
    const int tid = threadIdx.x;
    const int d   = tid & 63;
    const int j   = tid >> 6;

    float nuv[K];
#pragma unroll
    for (int q = 0; q < K; ++q) nuv[q] = nu[b * K + q];

    const float* g = negd + (size_t)b * M;
    float acc = 0.0f;
    const int NT = M / 256;             // 256 tiles

#if HAVE_TDM
    if (tid == 0) {
        tdm_load_f32(lds_offset_of(tiles), mem, 16384u, 1u, 16384u);
    }
#endif

    for (int t = 0; t < NT; ++t) {
        float* tile = tiles + (t & 1) * 16384;
#if HAVE_TDM
        if (tid == 0) {
            if (t + 1 < NT) {
                float* nxt = tiles + ((t + 1) & 1) * 16384;
                tdm_load_f32(lds_offset_of(nxt),
                             mem + (size_t)(t + 1) * 16384, 16384u, 1u, 16384u);
                __builtin_amdgcn_s_wait_tensorcnt(1);   // tile t complete
            } else {
                __builtin_amdgcn_s_wait_tensorcnt(0);
            }
        }
#else
#pragma unroll 8
        for (int i = 0; i < 64; ++i) {
            int idx = tid + i * 256;
            tile[idx] = mem[(size_t)t * 16384 + idx];
        }
#endif
        // stage weights (k-softmax column diffs)
        {
            float v  = g[t * 256 + tid];
            float y0 = fast_sigmoid(v + nuv[0]);
            float y1 = fast_sigmoid(v + nuv[1]);
            float y2 = fast_sigmoid(v + nuv[2]);
            float y3 = fast_sigmoid(v + nuv[3]);
            wl[tid * 4 + 0] = y0;
            wl[tid * 4 + 1] = y1 - y0;
            wl[tid * 4 + 2] = y2 - y1;
            wl[tid * 4 + 3] = y3 - y2;
        }
        __syncthreads();                // tile t + weights visible
#pragma unroll 8
        for (int m = 0; m < 256; ++m) {
            acc = fmaf(wl[m * 4 + j], tile[m * 64 + d], acc);
        }
        __syncthreads();                // done with tile t / wl
    }
    r0[((size_t)b * NDIM + d) * K + j] = acc;
}

// ---------------------------------------------------------------------------
// K5: one Hopfield step:  out[b,:,j] = softmax_m(2 r.mem_m - |mem_m|^2) @ mem
// ---------------------------------------------------------------------------
__global__ void hopfield_step_kernel(const float* __restrict__ r0,
                                     const float* __restrict__ mem,
                                     const float* __restrict__ mn,
                                     float* __restrict__ out) {
    __shared__ float rvec[NDIM];
    __shared__ float accum[NDIM];
    __shared__ float wred[8];
    __shared__ float bc;

    const int task = blockIdx.x;
    const int b    = task >> 2;
    const int j    = task & 3;
    const int tid  = threadIdx.x;
    const int lane = tid & 31;
    const int wave = tid >> 5;

    if (tid < NDIM) {
        rvec[tid]  = r0[((size_t)b * NDIM + tid) * K + j];
        accum[tid] = 0.0f;
    }
    __syncthreads();

    float lmax = -FLT_MAX;
    for (int m = tid; m < M; m += 256) {
        const float4* mr = (const float4*)(mem + (size_t)m * NDIM);
        __builtin_prefetch(mem + (size_t)(m + 256) * NDIM, 0, 0);
        float s = 0.0f;
#pragma unroll
        for (int q = 0; q < NDIM / 4; ++q) {
            float4 v = mr[q];
            s += rvec[4 * q + 0] * v.x + rvec[4 * q + 1] * v.y +
                 rvec[4 * q + 2] * v.z + rvec[4 * q + 3] * v.w;
        }
        lmax = fmaxf(lmax, 2.0f * s - mn[m]);
    }
#pragma unroll
    for (int off = 16; off >= 1; off >>= 1) lmax = fmaxf(lmax, __shfl_xor(lmax, off, 32));
    if (lane == 0) wred[wave] = lmax;
    __syncthreads();
    if (tid == 0) {
        float mx = wred[0];
#pragma unroll
        for (int w = 1; w < 8; ++w) mx = fmaxf(mx, wred[w]);
        bc = mx;
    }
    __syncthreads();
    const float T = bc;
    __syncthreads();

    float acc[NDIM];
#pragma unroll
    for (int q = 0; q < NDIM; ++q) acc[q] = 0.0f;
    float den = 0.0f;
    for (int m = tid; m < M; m += 256) {
        const float4* mr = (const float4*)(mem + (size_t)m * NDIM);
        float4 row[NDIM / 4];
        float s = 0.0f;
#pragma unroll
        for (int q = 0; q < NDIM / 4; ++q) {
            row[q] = mr[q];
            s += rvec[4 * q + 0] * row[q].x + rvec[4 * q + 1] * row[q].y +
                 rvec[4 * q + 2] * row[q].z + rvec[4 * q + 3] * row[q].w;
        }
        float w = __expf(2.0f * s - mn[m] - T);
        den += w;
#pragma unroll
        for (int q = 0; q < NDIM / 4; ++q) {
            acc[4 * q + 0] = fmaf(w, row[q].x, acc[4 * q + 0]);
            acc[4 * q + 1] = fmaf(w, row[q].y, acc[4 * q + 1]);
            acc[4 * q + 2] = fmaf(w, row[q].z, acc[4 * q + 2]);
            acc[4 * q + 3] = fmaf(w, row[q].w, acc[4 * q + 3]);
        }
    }
#pragma unroll
    for (int off = 16; off >= 1; off >>= 1) {
        den += __shfl_xor(den, off, 32);
#pragma unroll
        for (int q = 0; q < NDIM; ++q) acc[q] += __shfl_xor(acc[q], off, 32);
    }
    if (lane == 0) {
        wred[wave] = den;
#pragma unroll
        for (int q = 0; q < NDIM; ++q) atomicAdd(&accum[q], acc[q]);  // ds_add_f32
    }
    __syncthreads();
    if (tid < NDIM) {
        float dsum = 0.0f;
#pragma unroll
        for (int w = 0; w < 8; ++w) dsum += wred[w];
        out[((size_t)b * NDIM + tid) * K + j] = accum[tid] * __builtin_amdgcn_rcpf(dsum);
    }
}

// ---------------------------------------------------------------------------
extern "C" void kernel_launch(void* const* d_in, const int* in_sizes, int n_in,
                              void* d_out, int out_size, void* d_ws, size_t ws_size,
                              hipStream_t stream) {
    const float* x   = (const float*)d_in[0];   // (B, 64)
    const float* mem = (const float*)d_in[1];   // (M, 64)
    float* out = (float*)d_out;                 // (B, 64, 4)

    char* ws = (char*)d_ws;
    size_t o = 0;
    auto carve = [&](size_t bytes) -> char* {
        o = (o + 255) & ~(size_t)255;
        char* p = ws + o;
        o += bytes;
        return p;
    };
    _Float16* x16  = (_Float16*)carve((size_t)B * NDIM * sizeof(_Float16));
    _Float16* m16  = (_Float16*)carve((size_t)M * NDIM * sizeof(_Float16));
    float*    xn   = (float*)   carve((size_t)B * sizeof(float));
    float*    mn   = (float*)   carve((size_t)M * sizeof(float));
    float*    nu   = (float*)   carve((size_t)B * K * sizeof(float));
    float*    r0   = (float*)   carve((size_t)B * NDIM * K * sizeof(float));
    float*    negd = (float*)   carve((size_t)B * M * sizeof(float));     // 128 MB

    norms_kernel<<<(M + B + 255) / 256, 256, 0, stream>>>(x, mem, xn, mn);

    const int convN = M * NDIM + B * NDIM;
    tohalf_kernel<<<(convN + 255) / 256, 256, 0, stream>>>(x, mem, x16, m16);

    gemm_negd_kernel<<<(B / 16) * (M / 16) / 8, 256, 0, stream>>>(x16, m16, xn, mn, negd);

    const size_t smem_k3 = (size_t)(M + 32 * 5 + 8 + 32 * K) * sizeof(float);
    lml_bisect_kernel<<<B, 1024, smem_k3, stream>>>(negd, nu);

    const size_t smem_k4 = (size_t)(2 * 256 * 64 + 256 * 4) * sizeof(float);
    ksm_gemv_kernel<<<B, 256, smem_k4, stream>>>(negd, nu, mem, r0);

    hopfield_step_kernel<<<B * K, 256, 0, stream>>>(r0, mem, mn, out);
}
